// Decoder_9775345565829
// MI455X (gfx1250) — matrix operand order repaired and checked by reading, not compiled, not run on hardware
//
#include <hip/hip_runtime.h>
#include <hip/hip_bf16.h>
#include <math.h>

// ---------------- problem constants (from reference) ----------------
#define Bc   4096
#define Tc   20
#define Pc   20
#define Hc   256
#define Ec   128
#define Kc   5
#define CTXc 404          // E + H + P
#define G4   1024         // 4*H
#define NH   80           // heads output cols: 30+15+15+15 = 75, padded to 80

// ---------------- vector types for WMMA ----------------
typedef __attribute__((ext_vector_type(16))) _Float16 v16h;
typedef __attribute__((ext_vector_type(8)))  _Float16 v8h;
typedef __attribute__((ext_vector_type(8)))  float    v8f;

// ---------------- optional CDNA5 async global->LDS path ----------------
#if __has_builtin(__builtin_amdgcn_global_load_async_to_lds_b128) && \
    __has_builtin(__builtin_amdgcn_s_wait_asynccnt)
#define HAS_ASYNC_LDS 1
// exact parameter types per hipcc diagnostic:
//   param0: (vector_size(16) int) __device__ *   -> AS(1)
//   param1: same vector type in __shared__       -> AS(3)
typedef int v4i_ __attribute__((vector_size(4 * sizeof(int))));
typedef __attribute__((address_space(1))) v4i_ as1_v4i;
typedef __attribute__((address_space(3))) v4i_ as3_v4i;
#else
#define HAS_ASYNC_LDS 0
#endif

#if __has_builtin(__builtin_amdgcn_sched_group_barrier)
#define SCHED_GROUP(mask, cnt, id) __builtin_amdgcn_sched_group_barrier(mask, cnt, id)
#else
#define SCHED_GROUP(mask, cnt, id)
#endif

// =====================================================================
// Generic f16 WMMA GEMM:  C[M,N] = A[M,K] @ WT[N,K]^T (+ biases)
//   mode 0: C is f16,  C += bias_col[col]
//   mode 1: C is f32,  C += bias_col[col] (opt) + bias_row_full[row*N+col] (opt)
// Tile: 128(M) x 128(N) per block of 256 threads (8 waves).
// Per 64-K stage each wave: 2 sub-chunks x 2 groups of (4 B-frags -> 4 WMMAs),
// schedule pinned with sched_group_barrier so WMMAs issue back-to-back.
// A-fragments load global->VGPR directly (no cross-wave A reuse).
// W tile staged through padded LDS (bank-conflict-free fragment reads),
// filled via CDNA5 async DMA loads (ASYNCcnt) when available.
// =====================================================================
__global__ __launch_bounds__(256)
__attribute__((amdgpu_waves_per_eu(2)))
void gemm_f16_wmma(const _Float16* __restrict__ A,
                   const _Float16* __restrict__ WT,   // [N][K] row-major
                   _Float16* __restrict__ Cf16,
                   float*    __restrict__ Cf32,
                   const float* __restrict__ bias_col,
                   const float* __restrict__ bias_row_full,
                   int M, int N, int Kdim, int mode)
{
    __shared__ alignas(32) _Float16 Wlds[128][72];   // 64 K-halves + 8 pad

    const int tid   = threadIdx.x;
    const int wave  = tid >> 5;
    const int lane  = tid & 31;
    const int tileN = blockIdx.x * 128;
    const int tileM = blockIdx.y * 128;

    const int hs   = lane >> 4;       // half-select: 0 lanes 0-15, 1 lanes 16-31
    const int mrow = lane & 15;

    // this lane's A row (wave owns rows waveM..waveM+15; lane -> row mrow)
    const _Float16* Abase = A + (size_t)(tileM + wave * 16 + mrow) * Kdim;

    v8f acc[8] = {};

    const int nS = Kdim >> 6;         // 64-wide K stages
    for (int s = 0; s < nS; ++s) {
        const int k0 = s << 6;

        __syncthreads();              // protect Wlds against previous stage reads

        // ---- fill W tile: 128 rows x 64 halves = 1024 x 16B, 4 per thread ----
        #pragma unroll
        for (int it = 0; it < 4; ++it) {
            int i = tid + it * 256;          // 0..1023
            int r = i >> 3, q = i & 7;       // row, 16B-quad within row
            int n = tileN + r;
            if (n < N) {
#if HAS_ASYNC_LDS
                // CDNA5 async DMA: global -> LDS, tracked by ASYNCcnt
                __builtin_amdgcn_global_load_async_to_lds_b128(
                    (as1_v4i*)(WT + (size_t)n * Kdim + k0 + q * 8),
                    (as3_v4i*)&Wlds[r][q * 8], 0, 0);
#else
                *reinterpret_cast<uint4*>(&Wlds[r][q * 8]) =
                    *reinterpret_cast<const uint4*>(WT + (size_t)n * Kdim + k0 + q * 8);
#endif
            } else {
                *reinterpret_cast<uint4*>(&Wlds[r][q * 8]) = make_uint4(0u, 0u, 0u, 0u);
            }
        }
#if HAS_ASYNC_LDS
        __builtin_amdgcn_s_wait_asynccnt(0);
#endif
        // prefetch next stage of W into L2 (global_prefetch_b8)
        if (s + 1 < nS)
            __builtin_prefetch(WT + (size_t)(tileN + (tid >> 1)) * Kdim + k0 + 64, 0, 0);

        __syncthreads();

        // ---- A fragments for both 32-K sub-chunks: global -> VGPR, issued early ----
        // layout per ISA: K(i,lane) = (i>=8?16:0) + (hs?8:0) + (i&7)
        v16h af[2];
        #pragma unroll
        for (int kk = 0; kk < 2; ++kk) {
            const _Float16* ap = Abase + k0 + kk * 32 + hs * 8;
            v8h alo = *reinterpret_cast<const v8h*>(ap);
            v8h ahi = *reinterpret_cast<const v8h*>(ap + 16);
            #pragma unroll
            for (int i = 0; i < 8; ++i) { af[kk][i] = alo[i]; af[kk][i + 8] = ahi[i]; }
        }

        #pragma unroll
        for (int kk = 0; kk < 2; ++kk) {
            const int kb = kk * 32;
            #pragma unroll
            for (int g = 0; g < 2; ++g) {
                // ---- preload 4 B fragments, then burst 4 WMMAs ----
                v16h bf[4];
                #pragma unroll
                for (int c = 0; c < 4; ++c) {
                    const _Float16* br = &Wlds[(g * 4 + c) * 16 + mrow][kb + hs * 8];
                    v8h blo = *reinterpret_cast<const v8h*>(br);
                    v8h bhi = *reinterpret_cast<const v8h*>(br + 16);
                    #pragma unroll
                    for (int i = 0; i < 8; ++i) { bf[c][i] = blo[i]; bf[c][i + 8] = bhi[i]; }
                }
                #pragma unroll
                for (int c = 0; c < 4; ++c)
                    acc[g * 4 + c] = __builtin_amdgcn_wmma_f32_16x16x32_f16(
                                        false, af[kk], false, bf[c],
                                        (short)0, acc[g * 4 + c], false, false);
            }
        }

        // ---- pin the schedule: A-reads first, then 4x(8 ds-reads -> 4 WMMAs) ----
        SCHED_GROUP(0x020, 4, 0);     // 4 VMEM reads (A fragments)
        SCHED_GROUP(0x100, 8, 0);     // 8 DS reads  (4 B-frags)
        SCHED_GROUP(0x008, 4, 0);     // 4 WMMA
        SCHED_GROUP(0x100, 8, 0);
        SCHED_GROUP(0x008, 4, 0);
        SCHED_GROUP(0x100, 8, 0);
        SCHED_GROUP(0x008, 4, 0);
        SCHED_GROUP(0x100, 8, 0);
        SCHED_GROUP(0x008, 4, 0);
    }

    // ---- epilogue: D layout: vgpr r -> M = r + (lane<16?0:8), N = lane&15 ----
    const int mo   = (lane < 16) ? 0 : 8;
    const int colb = lane & 15;
    const int rowB = tileM + wave * 16;
    #pragma unroll
    for (int c = 0; c < 8; ++c) {
        int col = tileN + c * 16 + colb;
        if (col < N) {
            float bc = bias_col ? bias_col[col] : 0.0f;
            #pragma unroll
            for (int r = 0; r < 8; ++r) {
                int row = rowB + mo + r;
                float v = acc[c][r] + bc;
                if (bias_row_full) v += bias_row_full[(size_t)row * N + col];
                if (mode == 0) Cf16[(size_t)row * N + col] = (_Float16)v;
                else           Cf32[(size_t)row * N + col] = v;
            }
        }
    }
}

// =====================================================================
// Prep kernels (run once per launch)
// =====================================================================

// dst f16 [N][K] = transpose+cast of src f32 [K rows, ld stride]
__global__ void k_castT(const float* __restrict__ src, _Float16* __restrict__ dst,
                        int K, int N, int ld)
{
    int i = blockIdx.x * blockDim.x + threadIdx.x;
    if (i >= K * N) return;
    int n = i / K, k = i - n * K;
    dst[i] = (_Float16)src[(size_t)k * ld + n];
}

// WsumT[n][k] = f16( Wr[k][n] + Wk[128+k][n] )    (state_h == enc_h fold)
__global__ void k_wsumT(const float* __restrict__ Wr, const float* __restrict__ Wk,
                        _Float16* __restrict__ dst)
{
    int i = blockIdx.x * blockDim.x + threadIdx.x;
    if (i >= G4 * Hc) return;
    int n = i / Hc, k = i - n * Hc;
    dst[i] = (_Float16)(Wr[(size_t)k * G4 + n] + Wk[(size_t)(Ec + k) * G4 + n]);
}

// WheadsT[80][256]: cols 0..29 Wm | 30..44 Wy | 45..59 Wf | 60..74 Wfadj | pad 0
__global__ void k_headsT(const float* __restrict__ Wm, const float* __restrict__ Wy,
                         const float* __restrict__ Wf, const float* __restrict__ Wfa,
                         _Float16* __restrict__ dst)
{
    int i = blockIdx.x * blockDim.x + threadIdx.x;
    if (i >= NH * Hc) return;
    int n = i / Hc, k = i - n * Hc;
    float v = 0.0f;
    if      (n < 30) v = Wm [(size_t)k * 30 + n];
    else if (n < 45) v = Wy [(size_t)k * 15 + (n - 30)];
    else if (n < 60) v = Wf [(size_t)k * 15 + (n - 45)];
    else if (n < 75) v = Wfa[(size_t)k * 15 + (n - 60)];
    dst[i] = (_Float16)v;
}

__global__ void k_headbias(const float* __restrict__ bm, const float* __restrict__ by,
                           const float* __restrict__ bf, const float* __restrict__ bfa,
                           float* __restrict__ hb)
{
    int j = blockIdx.x * blockDim.x + threadIdx.x;
    if (j >= NH) return;
    float v = 0.0f;
    if      (j < 30) v = bm [j];
    else if (j < 45) v = by [j - 30];
    else if (j < 60) v = bf [j - 45];
    else if (j < 75) v = bfa[j - 60];
    hb[j] = v;
}

// ts_bias[t][j] = bl[j] + sum_{p<=t} Wk[384+p][j]
__global__ void k_tsbias(const float* __restrict__ Wk, const float* __restrict__ bl,
                         float* __restrict__ ts)
{
    int j = blockIdx.x * blockDim.x + threadIdx.x;
    if (j >= G4) return;
    float run = bl[j];
    #pragma unroll
    for (int t = 0; t < Tc; ++t) {
        run += Wk[(size_t)(Ec + Hc + t) * G4 + j];
        ts[(size_t)t * G4 + j] = run;
    }
}

__global__ void k_cast_f16(const float* __restrict__ src, _Float16* __restrict__ dst, int n)
{
    int i = blockIdx.x * blockDim.x + threadIdx.x;
    if (i < n) dst[i] = (_Float16)src[i];
}

// cond[b][c] = conditions[b, 0, c]
__global__ void k_init_cond(const float* __restrict__ conds, float* __restrict__ cond)
{
    int i = blockIdx.x * blockDim.x + threadIdx.x;
    if (i >= Bc * 5) return;
    int b = i / 5, c = i - b * 5;
    cond[i] = conds[(size_t)b * (Tc * 5) + c];
}

// =====================================================================
// Per-step kernels
// =====================================================================

// emb1[b][e] = f16( be1[e] + sum_{c<5} cond[b][c]*We1[c][e] )
__global__ void k_embed1(const float* __restrict__ cond, const float* __restrict__ We1,
                         const float* __restrict__ be1, _Float16* __restrict__ emb1)
{
    int i = blockIdx.x * blockDim.x + threadIdx.x;
    if (i >= Bc * Ec) return;
    int b = i >> 7, e = i & 127;
    const float* cb = cond + (size_t)b * 5;
    float acc = be1[e];
    #pragma unroll
    for (int c = 0; c < 5; ++c) acc += cb[c] * We1[c * Ec + e];
    emb1[i] = (_Float16)acc;
}

__device__ __forceinline__ float sigf(float x) { return 1.0f / (1.0f + expf(-x)); }

// gates: h = sig(o)*tanh( sig(f)*state_c + sig(i)*tanh(g) )
__global__ void k_gates(const float* __restrict__ z, const float* __restrict__ state_c,
                        _Float16* __restrict__ h16)
{
    int i = blockIdx.x * blockDim.x + threadIdx.x;
    if (i >= Bc * Hc) return;
    int b = i >> 8, j = i & 255;
    const float* zr = z + (size_t)b * G4;
    float ii = zr[j], ff = zr[j + 256], gg = zr[j + 512], oo = zr[j + 768];
    float cc = sigf(ff) * state_c[i] + sigf(ii) * tanhf(gg);
    h16[i] = (_Float16)(sigf(oo) * tanhf(cc));
}

// ---------------- deterministic hash RNG ----------------
__device__ __forceinline__ unsigned hmix(unsigned x)
{
    x ^= x >> 16; x *= 0x7feb352dU;
    x ^= x >> 15; x *= 0x846ca68bU;
    x ^= x >> 16; return x;
}
__device__ __forceinline__ float rng_u(int b, int t, int s)
{
    unsigned h = hmix((unsigned)b * 0x9E3779B1u ^
                      hmix((unsigned)t * 0x85EBCA77u ^ (unsigned)s * 0xC2B2AE3Du ^ 0x27D4EB2Fu));
    return ((float)(h >> 8) + 0.5f) * (1.0f / 16777216.0f);   // (0,1)
}
__device__ __forceinline__ float rng_n(int b, int t, int s)
{
    float u1 = rng_u(b, t, 2 * s), u2 = rng_u(b, t, 2 * s + 1);
    return sqrtf(-2.0f * logf(u1)) * cosf(6.2831853f * u2);
}
__device__ __forceinline__ int gumbel_argmax(const float* lg, int b, int t, int s)
{
    int best = 0; float bv = -1e30f;
    #pragma unroll
    for (int k = 0; k < 5; ++k) {
        float g = lg[k] - logf(-logf(rng_u(b, t, s * 8 + k)));
        if (g > bv) { bv = g; best = k; }
    }
    return best;
}
__device__ __forceinline__ void softmax5(const float* lg, float* a)
{
    float m = lg[0];
    #pragma unroll
    for (int k = 1; k < 5; ++k) m = fmaxf(m, lg[k]);
    float s = 0.0f;
    #pragma unroll
    for (int k = 0; k < 5; ++k) { a[k] = expf(lg[k] - m); s += a[k]; }
    float inv = 1.0f / s;
    #pragma unroll
    for (int k = 0; k < 5; ++k) a[k] *= inv;
}

// heads post-processing + sampling + output writes + next cond
__global__ void k_sample(const float* __restrict__ P, float* __restrict__ cond,
                         float* __restrict__ out, int t)
{
    int b = blockIdx.x * blockDim.x + threadIdx.x;
    if (b >= Bc) return;
    const float* p = P + (size_t)b * NH;

    float* om  = out;
    float* oy  = out + (size_t)Bc * Tc * 30;
    float* ofh = oy  + (size_t)Bc * Tc * 15;
    float* ofa = ofh + (size_t)Bc * Tc * 15;

    // ---- m head (30 outputs) ----
    float lg[5], a[5], mlo[5], slo[5], mla[5], sla[5], rho[5];
    #pragma unroll
    for (int k = 0; k < 5; ++k) {
        lg[k]  = p[k];
        mlo[k] = p[5 + k];   slo[k] = expf(p[10 + k]);
        mla[k] = p[15 + k];  sla[k] = expf(p[20 + k]);
        rho[k] = tanhf(p[25 + k]);
    }
    softmax5(lg, a);
    float* vm = om + ((size_t)b * Tc + t) * 30;
    #pragma unroll
    for (int k = 0; k < 5; ++k) {
        vm[k] = a[k]; vm[5 + k] = mlo[k]; vm[10 + k] = slo[k];
        vm[15 + k] = mla[k]; vm[20 + k] = sla[k]; vm[25 + k] = rho[k];
    }
    int im = gumbel_argmax(lg, b, t, 0);
    float z1 = rng_n(b, t, 100), z2 = rng_n(b, t, 101);
    float x_lo = mlo[im] + slo[im] * z1;
    float r = rho[im];
    float x_la = mla[im] + sla[im] * (r * z1 + sqrtf(fmaxf(0.0f, 1.0f - r * r)) * z2);

    // ---- y / f / fadj heads (15 outputs each) ----
    float sv[3];
    #pragma unroll
    for (int hh = 0; hh < 3; ++hh) {
        const float* q = p + 30 + hh * 15;
        float l2[5], al[5], mu[5], sg[5];
        #pragma unroll
        for (int k = 0; k < 5; ++k) { l2[k] = q[k]; mu[k] = q[5 + k]; sg[k] = expf(q[10 + k]); }
        softmax5(l2, al);
        float* dst = (hh == 0 ? oy : hh == 1 ? ofh : ofa) + ((size_t)b * Tc + t) * 15;
        #pragma unroll
        for (int k = 0; k < 5; ++k) { dst[k] = al[k]; dst[5 + k] = mu[k]; dst[10 + k] = sg[k]; }
        int ii = gumbel_argmax(l2, b, t, 10 + hh);
        sv[hh] = mu[ii] + sg[ii] * rng_n(b, t, 200 + hh);
    }

    float* cb = cond + (size_t)b * 5;
    cb[0] = x_lo; cb[1] = x_la; cb[2] = sv[0]; cb[3] = sv[1]; cb[4] = sv[2];
}

// =====================================================================
// Host launcher
// =====================================================================
static inline int cdiv(int a, int b) { return (a + b - 1) / b; }

extern "C" void kernel_launch(void* const* d_in, const int* in_sizes, int n_in,
                              void* d_out, int out_size, void* d_ws, size_t ws_size,
                              hipStream_t stream)
{
    (void)in_sizes; (void)n_in; (void)out_size; (void)ws_size;

    const float* conditions = (const float*)d_in[0];
    const float* state_h    = (const float*)d_in[1];
    const float* state_c    = (const float*)d_in[2];
    // d_in[3] = steps_n == Tc (compile-time)
    const float* We1 = (const float*)d_in[4];
    const float* be1 = (const float*)d_in[5];
    const float* We2 = (const float*)d_in[6];
    const float* be2 = (const float*)d_in[7];
    const float* Wk  = (const float*)d_in[8];
    const float* Wr  = (const float*)d_in[9];
    const float* bl  = (const float*)d_in[10];
    const float* Wm  = (const float*)d_in[11];
    const float* bm  = (const float*)d_in[12];
    const float* Wy  = (const float*)d_in[13];
    const float* by  = (const float*)d_in[14];
    const float* Wf  = (const float*)d_in[15];
    const float* bf  = (const float*)d_in[16];
    const float* Wfa = (const float*)d_in[17];
    const float* bfa = (const float*)d_in[18];

    // ---- workspace layout (256B aligned regions) ----
    char* w = (char*)d_ws; size_t off = 0;
    auto alloc = [&](size_t bytes) -> void* {
        void* p = w + off; off = (off + bytes + 255) & ~(size_t)255; return p;
    };
    _Float16* WkeT    = (_Float16*)alloc((size_t)G4 * Ec * 2);   // [1024][128]
    _Float16* We2T    = (_Float16*)alloc((size_t)Ec * Ec * 2);   // [128][128]
    _Float16* WsumT   = (_Float16*)alloc((size_t)G4 * Hc * 2);   // [1024][256]
    _Float16* WheadsT = (_Float16*)alloc((size_t)NH * Hc * 2);   // [80][256]
    float*    tsb     = (float*)   alloc((size_t)Tc * G4 * 4);   // [20][1024]
    float*    hb      = (float*)   alloc((size_t)NH * 4);
    _Float16* sh16    = (_Float16*)alloc((size_t)Bc * Hc * 2);   // state_h f16
    float*    hcbias  = (float*)   alloc((size_t)Bc * G4 * 4);   // 16 MB
    float*    condb   = (float*)   alloc((size_t)Bc * 5 * 4);
    _Float16* emb1    = (_Float16*)alloc((size_t)Bc * Ec * 2);
    _Float16* emb     = (_Float16*)alloc((size_t)Bc * Ec * 2);
    float*    zbuf    = (float*)   alloc((size_t)Bc * G4 * 4);   // 16 MB
    _Float16* h16     = (_Float16*)alloc((size_t)Bc * Hc * 2);
    float*    pbuf    = (float*)   alloc((size_t)Bc * NH * 4);

    // ---- one-time prep ----
    k_castT   <<<cdiv(Ec * G4, 256), 256, 0, stream>>>(Wk,  WkeT, Ec, G4, G4);
    k_castT   <<<cdiv(Ec * Ec, 256), 256, 0, stream>>>(We2, We2T, Ec, Ec, Ec);
    k_wsumT   <<<cdiv(G4 * Hc, 256), 256, 0, stream>>>(Wr, Wk, WsumT);
    k_headsT  <<<cdiv(NH * Hc, 256), 256, 0, stream>>>(Wm, Wy, Wf, Wfa, WheadsT);
    k_headbias<<<1, 128, 0, stream>>>(bm, by, bf, bfa, hb);
    k_tsbias  <<<cdiv(G4, 256), 256, 0, stream>>>(Wk, bl, tsb);
    k_cast_f16<<<cdiv(Bc * Hc, 256), 256, 0, stream>>>(state_h, sh16, Bc * Hc);
    k_init_cond<<<cdiv(Bc * 5, 256), 256, 0, stream>>>(conditions, condb);

    // hc_bias[B,1024] = state_h @ (Wr + Wk[128:384])    (WMMA, once)
    gemm_f16_wmma<<<dim3(G4 / 128, Bc / 128), 256, 0, stream>>>(
        sh16, WsumT, nullptr, hcbias, nullptr, nullptr, Bc, G4, Hc, 1);

    float* out = (float*)d_out;

    // ---- sequential decode: 20 steps ----
    for (int t = 0; t < Tc; ++t) {
        k_embed1<<<cdiv(Bc * Ec, 256), 256, 0, stream>>>(condb, We1, be1, emb1);

        // emb = emb1 @ We2 + be2  -> f16
        gemm_f16_wmma<<<dim3(1, Bc / 128), 256, 0, stream>>>(
            emb1, We2T, emb, nullptr, be2, nullptr, Bc, Ec, Ec, 0);

        // z = emb @ Wk[:128] + hc_bias + ts_bias[t]  -> f32
        gemm_f16_wmma<<<dim3(G4 / 128, Bc / 128), 256, 0, stream>>>(
            emb, WkeT, nullptr, zbuf, tsb + (size_t)t * G4, hcbias, Bc, G4, Ec, 1);

        k_gates<<<cdiv(Bc * Hc, 256), 256, 0, stream>>>(zbuf, state_c, h16);

        // p = h @ [Wm|Wy|Wf|Wfadj] + biases  -> f32 [B][80]
        gemm_f16_wmma<<<dim3(1, Bc / 128), 256, 0, stream>>>(
            h16, WheadsT, nullptr, pbuf, hb, nullptr, Bc, NH, Hc, 1);

        k_sample<<<cdiv(Bc, 256), 256, 0, stream>>>(pbuf, condb, out, t);
    }
}
